// MoELayer_16501264351883
// MI455X (gfx1250) — matrix operand order repaired and checked by reading, not compiled, not run on hardware
//
#include <hip/hip_runtime.h>
#include <stdint.h>

#define N_TOK 8192
#define DDIM  1024
#define NEXP  8

typedef __attribute__((ext_vector_type(16))) __bf16 v16bf;
typedef __attribute__((ext_vector_type(8)))  float  v8f;

union V16BF { uint4 q[2]; uint32_t u[8]; v16bf v; };

// Pack two fp32 into one dword of bf16 "major" halves (truncation) and one
// dword of bf16 "residual" halves. 3 VALU per element.
__device__ __forceinline__ void split_pair(float fe, float fo,
                                           uint32_t& hi, uint32_t& lo) {
    uint32_t ue = __float_as_uint(fe);
    uint32_t uo = __float_as_uint(fo);
    hi = __builtin_amdgcn_perm(uo, ue, 0x07060302u);      // {bf16(fo), bf16(fe)}
    float re = fe - __uint_as_float(ue & 0xffff0000u);
    float ro = fo - __uint_as_float(uo & 0xffff0000u);
    lo = __builtin_amdgcn_perm(__float_as_uint(ro), __float_as_uint(re),
                               0x07060302u);              // residual bf16 pair
}

// ---- gating: logits -> softmax -> top2 -> per-(expert,k) compacted lists ---
__global__ __launch_bounds__(256) void gate_kernel(
    const float* __restrict__ x, const float* __restrict__ gW,
    const float* __restrict__ gb,
    int* __restrict__ counts, int* __restrict__ lists, float* __restrict__ wlists)
{
    const int wave = threadIdx.x >> 5;
    const int lane = threadIdx.x & 31;
    const int n = blockIdx.x * 8 + wave;
    if (n >= N_TOK) return;

    float acc[8];
#pragma unroll
    for (int e = 0; e < 8; ++e) acc[e] = 0.0f;

    const float* xr = x + (size_t)n * DDIM;
    for (int d = lane; d < DDIM; d += 32) {
        float xv = xr[d];
        const float4* g4 = (const float4*)(gW + d * 8);
        float4 g0 = g4[0];
        float4 g1 = g4[1];
        acc[0] += xv * g0.x; acc[1] += xv * g0.y;
        acc[2] += xv * g0.z; acc[3] += xv * g0.w;
        acc[4] += xv * g1.x; acc[5] += xv * g1.y;
        acc[6] += xv * g1.z; acc[7] += xv * g1.w;
    }
#pragma unroll
    for (int off = 16; off > 0; off >>= 1)
#pragma unroll
        for (int e = 0; e < 8; ++e)
            acc[e] += __shfl_xor(acc[e], off, 32);

    if (lane == 0) {
        float l[8], p[8];
        float m = -3.4e38f;
#pragma unroll
        for (int e = 0; e < 8; ++e) { l[e] = acc[e] + gb[e]; m = fmaxf(m, l[e]); }
        float s = 0.0f;
#pragma unroll
        for (int e = 0; e < 8; ++e) { p[e] = __expf(l[e] - m); s += p[e]; }
        int i0 = 0;
#pragma unroll
        for (int e = 1; e < 8; ++e) if (p[e] > p[i0]) i0 = e;
        int i1 = (i0 == 0) ? 1 : 0;
#pragma unroll
        for (int e = 0; e < 8; ++e) if (e != i0 && p[e] > p[i1]) i1 = e;
        float inv = 1.0f / s;
        float w0 = p[i0] * inv, w1 = p[i1] * inv;

        int s0 = atomicAdd(&counts[i0 * 2 + 0], 1);
        lists [(i0 * 2 + 0) * N_TOK + s0] = n;
        wlists[(i0 * 2 + 0) * N_TOK + s0] = w0;
        int s1 = atomicAdd(&counts[i1 * 2 + 1], 1);
        lists [(i1 * 2 + 1) * N_TOK + s1] = n;
        wlists[(i1 * 2 + 1) * N_TOK + s1] = w1;
    }
}

// ---- per-expert gathered GEMM: 64 tokens x 128 cols per block --------------
// LDS holds hi/lo bf16-pair planes already in WMMA operand order:
//   A: Ahi/Alo[row 0..63][pk 0..15], row stride 20 dwords (pad, b128-aligned)
//   B: Bhi/Blo[pk 0..15][col 0..127], row stride 132 dwords (pad)
// pk = k/2; pair dword = {bf16(k_even) lo16, bf16(k_odd) hi16}.
__global__ __launch_bounds__(256) void moe_gemm_kernel(
    const float* __restrict__ x, const float* __restrict__ eW,
    const float* __restrict__ eb,
    const int* __restrict__ counts, const int* __restrict__ lists,
    const float* __restrict__ wlists, float* __restrict__ out, int phase)
{
    __shared__ uint32_t Ahi[64 * 20];
    __shared__ uint32_t Alo[64 * 20];
    __shared__ uint32_t Bhi[16 * 132];
    __shared__ uint32_t Blo[16 * 132];
    __shared__ int   tokS[64];
    __shared__ float wS[64];

    const int e   = blockIdx.z;
    const int cnt = counts[e * 2 + phase];
    const int rowbase = blockIdx.y * 64;
    if (rowbase >= cnt) return;                 // block-uniform: EXEC stays full
    const int colbase = blockIdx.x * 128;
    const int t = threadIdx.x;

    if (t < 64) {
        int g  = rowbase + t;
        int ok = (g < cnt);
        tokS[t] = ok ? lists [(e * 2 + phase) * N_TOK + g] : 0;
        wS[t]   = ok ? wlists[(e * 2 + phase) * N_TOK + g] : 0.0f;
    }
    __syncthreads();

    const int lane = t & 31;
    const int wave = t >> 5;
    const int h    = lane >> 4;        // lane half
    const int ln   = lane & 15;

    v8f acc[4];
#pragma unroll
    for (int s = 0; s < 4; ++s)
#pragma unroll
        for (int i = 0; i < 8; ++i) acc[s][i] = 0.0f;

    // staging assignments
    const int arow = t >> 2;           // 0..63, 4 threads/row
    const int aq   = t & 3;            // chunk of 8 consecutive k
    const int pkr  = t >> 4;           // 0..15 B pair-row
    const int c0   = (t & 15) * 8;     // 8 consecutive cols
    const int atok = tokS[arow];
    const float* arp = x + (size_t)atok * DDIM + aq * 8;
    const float* brp = eW + ((size_t)e * DDIM * DDIM) + colbase + c0;

    // consume bases (all DS offsets become immediates relative to these)
    const int aBase = ln * 20 + 4 * h;            // + sub*320 (+8 for 2nd chunk)
    const int bBase = wave * 16 + ln;             // + (j + (j&4) + 4h)*132

    for (int kk = 0; kk < DDIM / 32; ++kk) {
        const int kb = kk * 32;
        // ---- stage A: 8 consecutive k of one gathered row -> pair planes ----
        {
            const float4 a0 = *(const float4*)(arp + kb);
            const float4 a1 = *(const float4*)(arp + kb + 4);
            uint4 h4, l4;
            split_pair(a0.x, a0.y, h4.x, l4.x);
            split_pair(a0.z, a0.w, h4.y, l4.y);
            split_pair(a1.x, a1.y, h4.z, l4.z);
            split_pair(a1.z, a1.w, h4.w, l4.w);
            const int ai = arow * 20 + aq * 4;
            *(uint4*)&Ahi[ai] = h4;
            *(uint4*)&Alo[ai] = l4;
        }
        // ---- stage B: rows k=2*pkr,2*pkr+1, 8 cols -> pair planes ----------
        {
            const float* r0 = brp + (size_t)(kb + 2 * pkr) * DDIM;
            const float* r1 = r0 + DDIM;
            const float4 e0 = *(const float4*)r0;
            const float4 e1 = *(const float4*)(r0 + 4);
            const float4 o0 = *(const float4*)r1;
            const float4 o1 = *(const float4*)(r1 + 4);
            uint4 ha, hb, la, lb;
            split_pair(e0.x, o0.x, ha.x, la.x);
            split_pair(e0.y, o0.y, ha.y, la.y);
            split_pair(e0.z, o0.z, ha.z, la.z);
            split_pair(e0.w, o0.w, ha.w, la.w);
            split_pair(e1.x, o1.x, hb.x, lb.x);
            split_pair(e1.y, o1.y, hb.y, lb.y);
            split_pair(e1.z, o1.z, hb.z, lb.z);
            split_pair(e1.w, o1.w, hb.w, lb.w);
            const int bi = pkr * 132 + c0;
            *(uint4*)&Bhi[bi]     = ha;
            *(uint4*)&Bhi[bi + 4] = hb;
            *(uint4*)&Blo[bi]     = la;
            *(uint4*)&Blo[bi + 4] = lb;
        }
        __syncthreads();

        // ---- build operands: pure DS loads, no packing ----------------------
        V16BF bh, bl;
#pragma unroll
        for (int j = 0; j < 8; ++j) {
            bh.u[j] = Bhi[(j + (j & 4) + 4 * h) * 132 + bBase];
            bl.u[j] = Blo[(j + (j & 4) + 4 * h) * 132 + bBase];
        }
        V16BF ah[4], al[4];
#pragma unroll
        for (int sub = 0; sub < 4; ++sub) {
            const int ai = aBase + sub * 320;
            ah[sub].q[0] = *(const uint4*)&Ahi[ai];
            ah[sub].q[1] = *(const uint4*)&Ahi[ai + 8];
            al[sub].q[0] = *(const uint4*)&Alo[ai];
            al[sub].q[1] = *(const uint4*)&Alo[ai + 8];
        }
        // term-major: 4 independent WMMAs between reuses of each accumulator
#pragma unroll
        for (int sub = 0; sub < 4; ++sub)
            acc[sub] = __builtin_amdgcn_wmma_f32_16x16x32_bf16(
                false, ah[sub].v, false, bh.v, (short)0, acc[sub], false, false);
#pragma unroll
        for (int sub = 0; sub < 4; ++sub)
            acc[sub] = __builtin_amdgcn_wmma_f32_16x16x32_bf16(
                false, ah[sub].v, false, bl.v, (short)0, acc[sub], false, false);
#pragma unroll
        for (int sub = 0; sub < 4; ++sub)
            acc[sub] = __builtin_amdgcn_wmma_f32_16x16x32_bf16(
                false, al[sub].v, false, bh.v, (short)0, acc[sub], false, false);
        __syncthreads();
    }

    // ---- epilogue: out[tok, col] (phase 0 writes, phase 1 accumulates) -----
    const int col = colbase + wave * 16 + ln;
    const float bias = eb[e * DDIM + col];
#pragma unroll
    for (int sub = 0; sub < 4; ++sub) {
#pragma unroll
        for (int r = 0; r < 8; ++r) {
            const int ml = sub * 16 + r + 8 * h;   // C layout: VGPR r -> M = r + 8*half
            if (rowbase + ml < cnt) {
                const int   tok = tokS[ml];
                const float w   = wS[ml];
                const float v   = w * (acc[sub][r] + bias);
                float* o = out + (size_t)tok * DDIM + col;
                if (phase) *o += v; else *o = v;
            }
        }
    }
}

extern "C" void kernel_launch(void* const* d_in, const int* in_sizes, int n_in,
                              void* d_out, int out_size, void* d_ws, size_t ws_size,
                              hipStream_t stream) {
    (void)in_sizes; (void)n_in; (void)out_size; (void)ws_size;
    const float* x  = (const float*)d_in[0];
    const float* gW = (const float*)d_in[1];
    const float* gb = (const float*)d_in[2];
    const float* eW = (const float*)d_in[3];
    const float* eb = (const float*)d_in[4];
    float* out = (float*)d_out;

    // ws layout: [0,256) counts; then 16 lists of N_TOK ints; then 16 of N_TOK floats
    int*   counts = (int*)d_ws;
    int*   lists  = (int*)((char*)d_ws + 256);
    float* wlists = (float*)((char*)d_ws + 256 + (size_t)16 * N_TOK * sizeof(int));

    hipMemsetAsync(d_ws, 0, 256, stream);

    gate_kernel<<<dim3(N_TOK / 8), 256, 0, stream>>>(x, gW, gb, counts, lists, wlists);

    dim3 grid(DDIM / 128, N_TOK / 64, NEXP);   // cols, max row-tiles, experts
    moe_gemm_kernel<<<grid, 256, 0, stream>>>(x, eW, eb, counts, lists, wlists, out, 0);
    moe_gemm_kernel<<<grid, 256, 0, stream>>>(x, eW, eb, counts, lists, wlists, out, 1);
}
